// GraphClassifier_49005576847406
// MI455X (gfx1250) — matrix operand order repaired and checked by reading, not compiled, not run on hardware
//
#include <hip/hip_runtime.h>
#include <hip/hip_bf16.h>

#define NN 4096
#define IN_DIM 256
#define HID 128
#define HEADS 4
#define DH 32
#define LAYERS 2
#define CLASSES 10

typedef __attribute__((ext_vector_type(16))) _Float16 v16h;
typedef __attribute__((ext_vector_type(8)))  _Float16 v8h;
typedef __attribute__((ext_vector_type(8)))  float    v8f;

// ---- gfx1250 async global->LDS helpers -------------------------------------
__device__ __forceinline__ void async_load_b128(unsigned lds_off, const _Float16* gptr) {
    unsigned long long ga = (unsigned long long)(uintptr_t)gptr;
    asm volatile("global_load_async_to_lds_b128 %0, %1, off"
                 :: "v"(lds_off), "v"(ga) : "memory");
}
__device__ __forceinline__ void wait_async_le2() {
    asm volatile("s_wait_asynccnt 0x2" ::: "memory");
}
__device__ __forceinline__ void wait_async_le0() {
    asm volatile("s_wait_asynccnt 0x0" ::: "memory");
}

// ---------------------------------------------------------------- mask pack
__global__ __launch_bounds__(256) void pack_mask_kernel(const int* __restrict__ adj,
                                                        unsigned* __restrict__ bits) {
    int w = blockIdx.x * blockDim.x + threadIdx.x;      // word index, N*N/32 total
    int n  = w >> 7;                                    // row (128 words / row)
    int wc = w & 127;
    const int* p = adj + (size_t)n * NN + wc * 32;
    unsigned v = 0;
#pragma unroll
    for (int j = 0; j < 32; ++j) v |= (p[j] != 0 ? 1u : 0u) << j;
    bits[w] = v;
}

// ------------------------------------------------- input projection + L2 norm
__global__ __launch_bounds__(128) void proj_in_kernel(const float* __restrict__ x,
                                                      const float* __restrict__ W_in,
                                                      const float* __restrict__ b_in,
                                                      float* __restrict__ h) {
    __shared__ float xs[IN_DIM];
    __shared__ float red[HID];
    int row = blockIdx.x, d = threadIdx.x;
    xs[d]       = x[(size_t)row * IN_DIM + d];
    xs[d + 128] = x[(size_t)row * IN_DIM + d + 128];
    __syncthreads();
    float acc = b_in[d];
#pragma unroll 8
    for (int i = 0; i < IN_DIM; ++i) acc = fmaf(xs[i], W_in[(size_t)i * HID + d], acc);
    red[d] = acc * acc;
    __syncthreads();
    for (int s = 64; s > 0; s >>= 1) {
        if (d < s) red[d] += red[d + s];
        __syncthreads();
    }
    float scale = 1.0f / fmaxf(sqrtf(red[0]), 1e-12f);
    h[(size_t)row * HID + d] = acc * scale;
}

// ------------------------------------------- QKV projection (f16, Vt transposed)
__global__ __launch_bounds__(128) void qkv_kernel(const float* __restrict__ h,
                                                  const float* __restrict__ Wq, const float* __restrict__ bq,
                                                  const float* __restrict__ Wk, const float* __restrict__ bk,
                                                  const float* __restrict__ Wv, const float* __restrict__ bv,
                                                  _Float16* __restrict__ q,
                                                  _Float16* __restrict__ k,
                                                  _Float16* __restrict__ vt) {
    __shared__ float hs[HID];
    int row = blockIdx.x, d = threadIdx.x;
    hs[d] = h[(size_t)row * HID + d];
    __syncthreads();
    float aq = bq[d], ak = bk[d], av = bv[d];
#pragma unroll 8
    for (int i = 0; i < HID; ++i) {
        float hv = hs[i];
        aq = fmaf(hv, Wq[(size_t)i * HID + d], aq);
        ak = fmaf(hv, Wk[(size_t)i * HID + d], ak);
        av = fmaf(hv, Wv[(size_t)i * HID + d], av);
    }
    q[(size_t)row * HID + d] = (_Float16)(aq * 0.17677669529663687f); // 1/sqrt(32) baked in
    k[(size_t)row * HID + d] = (_Float16)ak;
    vt[(size_t)d * NN + row] = (_Float16)av;                           // transposed V
}

// --------------------------------------------- flash attention, WMMA f16->f32
// grid: (NN/(16*4), HEADS), block: 128 (4 waves, one 16-query tile per wave).
// K/V chunks are staged cooperatively into double-buffered LDS with
// global_load_async_to_lds_b128 and shared by all 4 waves of the block.
__global__ __launch_bounds__(128) void attn_kernel(const _Float16* __restrict__ q,
                                                   const _Float16* __restrict__ k,
                                                   const _Float16* __restrict__ vt,
                                                   const unsigned* __restrict__ maskbits,
                                                   float* __restrict__ out) {
    __shared__ _Float16 kbuf[2][32][32];   // [buf][key_local][dh]
    __shared__ _Float16 vbuf[2][32][32];   // [buf][dh][key_local]
    __shared__ _Float16 psmem_all[4][16 * 32];

    const int tid  = threadIdx.x;
    const int lane = tid & 31;
    const int wave = tid >> 5;
    const int head = blockIdx.y;
    const int qbase = (blockIdx.x * 4 + wave) * 16;
    _Float16* psmem = psmem_all[wave];

    const int lh = lane & 15;   // 0..15
    const int lg = lane >> 4;   // 0 or 1

    // ---- cooperative staging assignment: 128 threads x 16B cover one 2KB tile
    const int srow  = tid >> 2;          // 0..31
    const int spart = (tid & 3) * 8;     // half-offset within a 32-half row
    const unsigned k_lds0 = (unsigned)(uintptr_t)&kbuf[0][srow][spart];
    const unsigned v_lds0 = (unsigned)(uintptr_t)&vbuf[0][srow][spart];
    const unsigned buf_bytes = 32 * 32 * 2;
    const _Float16* kg = k  + (size_t)srow * HID + head * DH + spart;          // + kb*HID
    const _Float16* vg = vt + (size_t)(head * DH + srow) * NN + spart;         // + kb

    // ---- Q tile (16x32) in A-matrix layout: lane: m=lh, K=lg*16+j contiguous
    const _Float16* qrow = q + (size_t)(qbase + lh) * HID + head * DH;
    v8h q_lo = *(const v8h*)(qrow + lg * 16);
    v8h q_hi = *(const v8h*)(qrow + lg * 16 + 8);
    v16h a_q;
#pragma unroll
    for (int j = 0; j < 8; ++j) { a_q[j] = q_lo[j]; a_q[8 + j] = q_hi[j]; }

    float M[8], L[8];
    v8f O0 = {}, O1 = {};
#pragma unroll
    for (int r = 0; r < 8; ++r) { M[r] = -3.0e38f; L[r] = 0.0f; }

    // prologue: prefetch chunk 0 into buffer 0
    async_load_b128(k_lds0, kg);
    async_load_b128(v_lds0, vg);

    for (int c = 0; c < NN / 32; ++c) {
        const int kb = c * 32;
        if (c + 1 < NN / 32) {
            const unsigned nb = (unsigned)((c + 1) & 1) * buf_bytes;
            async_load_b128(k_lds0 + nb, kg + (size_t)(kb + 32) * HID);
            async_load_b128(v_lds0 + nb, vg + (kb + 32));
            wait_async_le2();            // chunk c landed; chunk c+1 in flight
        } else {
            wait_async_le0();
        }
        __syncthreads();                 // all waves' segments visible

        const _Float16 (*kc)[32] = kbuf[c & 1];
        const _Float16 (*vc)[32] = vbuf[c & 1];

        // ---- K chunk as two B matrices (32x16): lane: n=key(lh), K=dh lg*16+j contiguous
        v8h k0a = *(const v8h*)&kc[lh][lg * 16];
        v8h k0b = *(const v8h*)&kc[lh][lg * 16 + 8];
        v8h k1a = *(const v8h*)&kc[16 + lh][lg * 16];
        v8h k1b = *(const v8h*)&kc[16 + lh][lg * 16 + 8];
        v16h b0, b1;
#pragma unroll
        for (int j = 0; j < 8; ++j) {
            b0[j] = k0a[j]; b0[8 + j] = k0b[j];
            b1[j] = k1a[j]; b1[8 + j] = k1b[j];
        }

        v8f cz = {};
        v8f s0 = __builtin_amdgcn_wmma_f32_16x16x32_f16(false, a_q, false, b0, (short)0, cz, false, false);
        v8f s1 = __builtin_amdgcn_wmma_f32_16x16x32_f16(false, a_q, false, b1, (short)0, cz, false, false);

        // ---- mask + online softmax (per-lane rows m = r + 8*lg, col n = lh)
        const int word = kb >> 5;
#pragma unroll
        for (int r = 0; r < 8; ++r) {
            int m = qbase + r + 8 * lg;
            unsigned mw = maskbits[(size_t)m * (NN / 32) + word];
            float v0 = ((mw >> lh) & 1u)        ? s0[r] : -1.0e9f;
            float v1 = ((mw >> (lh + 16)) & 1u) ? s1[r] : -1.0e9f;
            float cm = fmaxf(v0, v1);
#pragma unroll
            for (int off = 1; off < 16; off <<= 1) cm = fmaxf(cm, __shfl_xor(cm, off));
            float Mn = fmaxf(M[r], cm);
            float al = __expf(M[r] - Mn);
            float p0 = __expf(v0 - Mn);
            float p1 = __expf(v1 - Mn);
            float ls = p0 + p1;
#pragma unroll
            for (int off = 1; off < 16; off <<= 1) ls += __shfl_xor(ls, off);
            L[r] = L[r] * al + ls;
            M[r] = Mn;
            O0[r] *= al;
            O1[r] *= al;
            // stash exp'd probs in LDS, row-major [16 rows x 32 keys] (f16)
            psmem[(r + 8 * lg) * 32 + lh]      = (_Float16)p0;
            psmem[(r + 8 * lg) * 32 + 16 + lh] = (_Float16)p1;
        }

        // ---- reload P in A-matrix layout (same wave -> LDS in order, no barrier)
        const _Float16* pp = psmem + lh * 32 + lg * 16;
        v8h plo = *(const v8h*)pp;
        v8h phi = *(const v8h*)(pp + 8);
        v16h a_p;
#pragma unroll
        for (int j = 0; j < 8; ++j) { a_p[j] = plo[j]; a_p[8 + j] = phi[j]; }

        // ---- V chunk as two B matrices: lane: n=dh(lh / 16+lh), K=key lg*16+j contiguous
        v8h v0a = *(const v8h*)&vc[lh][lg * 16];
        v8h v0b = *(const v8h*)&vc[lh][lg * 16 + 8];
        v8h v1a = *(const v8h*)&vc[16 + lh][lg * 16];
        v8h v1b = *(const v8h*)&vc[16 + lh][lg * 16 + 8];
        v16h bv0, bv1;
#pragma unroll
        for (int j = 0; j < 8; ++j) {
            bv0[j] = v0a[j]; bv0[8 + j] = v0b[j];
            bv1[j] = v1a[j]; bv1[8 + j] = v1b[j];
        }

        O0 = __builtin_amdgcn_wmma_f32_16x16x32_f16(false, a_p, false, bv0, (short)0, O0, false, false);
        O1 = __builtin_amdgcn_wmma_f32_16x16x32_f16(false, a_p, false, bv1, (short)0, O1, false, false);

        __syncthreads();                 // protect buffer reuse two iterations ahead
    }

    // ---- finalize: O / L, scatter to [N, HID]
#pragma unroll
    for (int r = 0; r < 8; ++r) {
        int m = qbase + r + 8 * lg;
        float inv = 1.0f / L[r];
        out[(size_t)m * HID + head * DH + lh]      = O0[r] * inv;
        out[(size_t)m * HID + head * DH + 16 + lh] = O1[r] * inv;
    }
}

// --------------------------------------- O-projection + ReLU + residual (in h)
__global__ __launch_bounds__(128) void oproj_kernel(const float* __restrict__ attn,
                                                    const float* __restrict__ Wo,
                                                    const float* __restrict__ bo,
                                                    float* __restrict__ h) {
    __shared__ float as[HID];
    int row = blockIdx.x, d = threadIdx.x;
    as[d] = attn[(size_t)row * HID + d];
    __syncthreads();
    float acc = bo[d];
#pragma unroll 8
    for (int i = 0; i < HID; ++i) acc = fmaf(as[i], Wo[(size_t)i * HID + d], acc);
    float y = fmaxf(acc, 0.0f);
    size_t idx = (size_t)row * HID + d;
    h[idx] = 0.5f * h[idx] + 0.5f * y;
}

// ------------------------------------------------------------------- final FC
__global__ __launch_bounds__(32) void fc_kernel(const float* __restrict__ h,
                                                const float* __restrict__ W_fc,
                                                const float* __restrict__ b_fc,
                                                float* __restrict__ out) {
    int row = blockIdx.x, c = threadIdx.x;
    if (c < CLASSES) {
        float acc = b_fc[c];
#pragma unroll 8
        for (int i = 0; i < HID; ++i)
            acc = fmaf(h[(size_t)row * HID + i], W_fc[(size_t)i * CLASSES + c], acc);
        out[(size_t)row * CLASSES + c] = acc;
    }
}

extern "C" void kernel_launch(void* const* d_in, const int* in_sizes, int n_in,
                              void* d_out, int out_size, void* d_ws, size_t ws_size,
                              hipStream_t stream) {
    const float* x     = (const float*)d_in[0];
    const int*   adj   = (const int*)d_in[1];
    const float* W_in  = (const float*)d_in[2];
    const float* b_in  = (const float*)d_in[3];
    const float* Wq    = (const float*)d_in[4];
    const float* bq    = (const float*)d_in[5];
    const float* Wk    = (const float*)d_in[6];
    const float* bk    = (const float*)d_in[7];
    const float* Wv    = (const float*)d_in[8];
    const float* bv    = (const float*)d_in[9];
    const float* Wo    = (const float*)d_in[10];
    const float* bo    = (const float*)d_in[11];
    const float* W_fc  = (const float*)d_in[12];
    const float* b_fc  = (const float*)d_in[13];
    float* out = (float*)d_out;

    char* ws = (char*)d_ws;
    unsigned*  maskbits = (unsigned*)(ws);                  // 2 MB
    float*     h        = (float*)(ws + (2u << 20));        // 2 MB
    float*     attn     = (float*)(ws + (4u << 20));        // 2 MB
    _Float16*  qb       = (_Float16*)(ws + (6u << 20));     // 1 MB
    _Float16*  kb       = (_Float16*)(ws + (7u << 20));     // 1 MB
    _Float16*  vtb      = (_Float16*)(ws + (8u << 20));     // 1 MB

    pack_mask_kernel<<<(NN * (NN / 32)) / 256, 256, 0, stream>>>(adj, maskbits);
    proj_in_kernel<<<NN, 128, 0, stream>>>(x, W_in, b_in, h);

    for (int l = 0; l < LAYERS; ++l) {
        const size_t wOff = (size_t)l * HID * HID;
        const size_t bOff = (size_t)l * HID;
        qkv_kernel<<<NN, 128, 0, stream>>>(h, Wq + wOff, bq + bOff, Wk + wOff, bk + bOff,
                                           Wv + wOff, bv + bOff, qb, kb, vtb);
        attn_kernel<<<dim3(NN / 64, HEADS), 128, 0, stream>>>(qb, kb, vtb, maskbits, attn);
        oproj_kernel<<<NN, 128, 0, stream>>>(attn, Wo + wOff, bo + bOff, h);
    }

    fc_kernel<<<NN, 32, 0, stream>>>(h, W_fc, b_fc, out);
}
